// RecurrentResonanceModel_86861418594374
// MI455X (gfx1250) — matrix operand order repaired and checked by reading, not compiled, not run on hardware
//
#include <hip/hip_runtime.h>
#include <hip/hip_bf16.h>
#include <cstdint>

// ---- problem constants -------------------------------------------------
#define N_EVENTS   16
#define N_ATOMS    512
#define LATENT     128
#define RES        32768
#define N_FRAMES   128
#define CHUNK      256        // samples handled by one workgroup
#define KS         64         // atoms (K) per LDS stage
#define NSLICE     (N_ATOMS / KS)  // 8
#define LDB        72         // ldsT row stride in ushort (16B aligned rows)
#define LDA        72         // ldsA row stride in ushort
#define WINDOWED_ELEMS (N_EVENTS * RES)              // 524288
#define NEWMOM_ELEMS   (N_EVENTS * N_ATOMS * N_FRAMES) // 1048576

typedef __attribute__((ext_vector_type(16))) __bf16       v16bf;
typedef __attribute__((ext_vector_type(8)))  float        v8f;
typedef __attribute__((ext_vector_type(4)))  unsigned int u32x4;
typedef __attribute__((__vector_size__(4 * sizeof(int)))) int v4i;

union FragB16 { v16bf v; u32x4 q[2]; };

#define AS1 __attribute__((address_space(1)))
#define AS3 __attribute__((address_space(3)))

#if __has_builtin(__builtin_amdgcn_global_load_async_to_lds_b128)
#define HAVE_ASYNC_LDS 1
#else
#define HAVE_ASYNC_LDS 0
#endif

__device__ __forceinline__ void async_copy16(const void* g, void* l) {
#if HAVE_ASYNC_LDS
  __builtin_amdgcn_global_load_async_to_lds_b128((v4i*)g, (AS3 v4i*)l, 0, 0);
#endif
}

__device__ __forceinline__ void wait_async0() {
#if HAVE_ASYNC_LDS
#if __has_builtin(__builtin_amdgcn_s_wait_asynccnt)
  __builtin_amdgcn_s_wait_asynccnt(0);
#else
  asm volatile("s_wait_asynccnt 0x0" ::: "memory");
#endif
#endif
}

__device__ __forceinline__ unsigned short f2bf(float f) {
  unsigned u = __float_as_uint(f);
  u += 0x7fffu + ((u >> 16) & 1u);          // round-to-nearest-even
  return (unsigned short)(u >> 16);
}

// ---- kernel 1: per-(event,atom) amps: new_mom output + bf16 frame amps ----
// grid: 32 x 256.  Each block handles one event b, 256 atoms.
__global__ __launch_bounds__(256)
void frames_kernel(const float* __restrict__ x,
                   const float* __restrict__ Wi, const float* __restrict__ bi,
                   const float* __restrict__ Wm, const float* __restrict__ bm,
                   float* __restrict__ nm_out,           // [16][512][128]
                   unsigned short* __restrict__ ampF,    // [128][16][512] bf16
                   unsigned int* __restrict__ maxbuf)    // [16]
{
  __shared__ float xs[LATENT];
  const int b = blockIdx.x >> 1;
  const int a = ((blockIdx.x & 1) << 8) + threadIdx.x;
  if (threadIdx.x < LATENT) xs[threadIdx.x] = x[b * LATENT + threadIdx.x];
  if (blockIdx.x == 0 && threadIdx.x < N_EVENTS) maxbuf[threadIdx.x] = 0u;
  __syncthreads();

  const float4* wi4 = (const float4*)(Wi + (size_t)a * LATENT);
  const float4* wm4 = (const float4*)(Wm + (size_t)a * LATENT);
  const float4* xv4 = (const float4*)xs;
  float acc1 = 0.f, acc2 = 0.f;
#pragma unroll
  for (int i = 0; i < LATENT / 4; ++i) {
    float4 xv = xv4[i];
    float4 w1 = wi4[i];
    float4 w2 = wm4[i];
    acc1 += xv.x * w1.x + xv.y * w1.y + xv.z * w1.z + xv.w * w1.w;
    acc2 += xv.x * w2.x + xv.y * w2.y + xv.z * w2.z + xv.w * w2.w;
  }
  const float initial = fmaxf(acc1 + bi[a], 0.0f);
  const float t = acc2 + bm[a];
  const float mom = 0.02f + 0.882f / (1.0f + __expf(-t));   // base + sigmoid*res_factor

  float* nm = nm_out + ((size_t)(b * N_ATOMS + a)) * N_FRAMES;
  float p = 1.0f;
  for (int f = 0; f < N_FRAMES; ++f) {
    p *= mom;                                  // mom^(f+1) == exp(log(mom)*step)
    nm[f] = p;
    ampF[((size_t)f * N_EVENTS + b) * N_ATOMS + a] = f2bf(p * initial);
  }
}

// ---- kernel 2: per-chunk GEMM (M=48, N=256, K=512) + interp blend --------
// grid: 128 x 256 (8 waves).  Chunk g covers samples [256g, 256g+256).
__global__ __launch_bounds__(256)
void chunk_gemm_kernel(const float* __restrict__ atoms,          // [512][32768]
                       const unsigned short* __restrict__ ampF,  // [128][16][512] bf16
                       float* __restrict__ pre,                  // [16][32768] un-normalized
                       unsigned int* __restrict__ maxbuf)        // [16]
{
  __shared__ __align__(16) unsigned short ldsT[CHUNK * LDB];   // [sample][k] bf16
  __shared__ __align__(16) unsigned short ldsA[48 * LDA];      // [row][k]   bf16
  __shared__ unsigned int smax[N_EVENTS];

  const int g  = blockIdx.x;
  const int s0 = g * CHUNK;
  const int f0 = (g > 0) ? (g - 1) : 0;            // frame row-block 0
  const int f1 = g;                                 // frame row-block 1
  const int f2 = (g < N_FRAMES - 1) ? (g + 1) : (N_FRAMES - 1); // row-block 2

  const int tid  = threadIdx.x;
  const int lane = tid & 31;
  const int wv   = tid >> 5;
  const int hi   = lane >> 4;
  const int col  = lane & 15;

  if (tid < N_EVENTS) smax[tid] = 0u;

  v8f acc[3][2];
  v8f vzero = {};
#pragma unroll
  for (int mt = 0; mt < 3; ++mt)
#pragma unroll
    for (int j = 0; j < 2; ++j) acc[mt][j] = vzero;

  const int krow = tid >> 6;          // 0..3
  const int scol = (tid & 63) << 2;   // 0..252 step 4

  for (int sl = 0; sl < NSLICE; ++sl) {
    const int ks = sl * KS;
    __syncthreads();

    // prefetch next K-slice of atoms into L2 while we fill LDS for this one
    if (sl + 1 < NSLICE) {
      const float* nxt = atoms + (size_t)(ks + KS + (tid >> 2)) * RES + s0 + ((tid & 3) << 6);
      __builtin_prefetch(nxt, 0, 1);
    }

#if HAVE_ASYNC_LDS
    // stage A rows via async VMEM->LDS: 48 rows x KS ushorts = 384 x 16B packets
    {
      int c = tid;
#pragma unroll
      for (int rep = 0; rep < 2; ++rep, c += 256) {
        if (c < 48 * KS / 8) {
          const int r  = c >> 3;            // 0..47
          const int kg = (c & 7) << 3;      // 0..56 step 8
          const int fi = r >> 4;
          const int fb = (fi == 0) ? f0 : ((fi == 1) ? f1 : f2);
          const int ev = r & 15;
          const unsigned short* gsrc =
              ampF + ((size_t)fb * N_EVENTS + ev) * N_ATOMS + ks + kg;
          async_copy16(gsrc, ldsA + r * LDA + kg);
        }
      }
    }
#else
    // stage A rows: 3 frames x 16 events x KS atoms -> ldsA (VGPR round-trip)
    for (int e = tid; e < 48 * KS; e += 256) {
      const int r  = e >> 6;          // 0..47 (KS==64)
      const int k  = e & (KS - 1);
      const int fi = r >> 4;
      const int fb = (fi == 0) ? f0 : ((fi == 1) ? f1 : f2);
      const int ev = r & 15;
      ldsA[r * LDA + k] = ampF[((size_t)fb * N_EVENTS + ev) * N_ATOMS + ks + k];
    }
#endif

    // stage atoms[ks..ks+64)[s0..s0+256) -> ldsT (bf16, sample-major)
#pragma unroll
    for (int it = 0; it < KS / 4; ++it) {
      const int k = it * 4 + krow;
      const float4 av = *(const float4*)(atoms + (size_t)(ks + k) * RES + s0 + scol);
      ldsT[(scol + 0) * LDB + k] = f2bf(av.x);
      ldsT[(scol + 1) * LDB + k] = f2bf(av.y);
      ldsT[(scol + 2) * LDB + k] = f2bf(av.z);
      ldsT[(scol + 3) * LDB + k] = f2bf(av.w);
    }

    wait_async0();
    __syncthreads();

    // compute: wave wv owns samples [wv*32, wv*32+32) -> 2 N-tiles x 3 M-tiles
#pragma unroll
    for (int kk = 0; kk < KS; kk += 32) {
      FragB16 bfr[2];
#pragma unroll
      for (int j = 0; j < 2; ++j) {
        const unsigned short* bp = ldsT + (wv * 32 + j * 16 + col) * LDB + kk + hi * 16;
        bfr[j].q[0] = *(const u32x4*)(bp);
        bfr[j].q[1] = *(const u32x4*)(bp + 8);
      }
      FragB16 afr[3];
#pragma unroll
      for (int mt = 0; mt < 3; ++mt) {
        const unsigned short* ap = ldsA + (mt * 16 + col) * LDA + kk + hi * 16;
        afr[mt].q[0] = *(const u32x4*)(ap);
        afr[mt].q[1] = *(const u32x4*)(ap + 8);
      }
#pragma unroll
      for (int mt = 0; mt < 3; ++mt)
#pragma unroll
        for (int j = 0; j < 2; ++j)
          acc[mt][j] = __builtin_amdgcn_wmma_f32_16x16x32_bf16(
              false, afr[mt].v, false, bfr[j].v, (short)0, acc[mt][j], false, false);
    }
  }

  // epilogue: linear-interp blend across the three frame row-blocks
#pragma unroll
  for (int j = 0; j < 2; ++j) {
    const int s_l = wv * 32 + j * 16 + col;
    const int gs  = s0 + s_l;
    float src  = ((float)gs + 0.5f) * (1.0f / 256.0f) - 0.5f;
    float srcc = fminf(fmaxf(src, 0.0f), 127.0f);
    int   x0   = (int)floorf(srcc);
    if (x0 > N_FRAMES - 1) x0 = N_FRAMES - 1;
    const int   x1  = (x0 + 1 < N_FRAMES) ? (x0 + 1) : (N_FRAMES - 1);
    const float wgt = srcc - (float)x0;
    const int r0 = x0 - (g - 1);   // 0..2
    const int r1 = x1 - (g - 1);   // 0..2

    v8f P = (r0 == 0) ? acc[0][j] : ((r0 == 1) ? acc[1][j] : acc[2][j]);
    v8f Q = (r1 == 0) ? acc[0][j] : ((r1 == 1) ? acc[1][j] : acc[2][j]);
    v8f R = P * (1.0f - wgt) + Q * wgt;

#pragma unroll
    for (int v = 0; v < 8; ++v) {
      const int m = v + 8 * hi;     // event index
      __builtin_nontemporal_store(R[v], &pre[(size_t)m * RES + gs]);
      atomicMax(&smax[m], __float_as_uint(fabsf(R[v])));
    }
  }
  __syncthreads();
  if (tid < N_EVENTS) atomicMax(&maxbuf[tid], smax[tid]);
}

// ---- kernel 3: per-event max-norm ---------------------------------------
__global__ __launch_bounds__(256)
void normalize_kernel(const float* __restrict__ pre,
                      const unsigned int* __restrict__ maxbuf,
                      float* __restrict__ out)
{
  const int i = blockIdx.x * 256 + threadIdx.x;
  if (i < WINDOWED_ELEMS) {
    const int b = i >> 15;   // RES = 2^15
    const float mx = __uint_as_float(maxbuf[b]);
    const float v = __builtin_nontemporal_load(&pre[i]);
    out[i] = v / (mx + 1e-8f);
  }
}

// ---- launcher ------------------------------------------------------------
extern "C" void kernel_launch(void* const* d_in, const int* in_sizes, int n_in,
                              void* d_out, int out_size, void* d_ws, size_t ws_size,
                              hipStream_t stream) {
  const float* x     = (const float*)d_in[0];
  const float* Wi    = (const float*)d_in[1];
  const float* bi    = (const float*)d_in[2];
  const float* Wm    = (const float*)d_in[3];
  const float* bm    = (const float*)d_in[4];
  const float* atoms = (const float*)d_in[5];

  float* out    = (float*)d_out;                 // [16][32768] windowed
  float* nm_out = out + WINDOWED_ELEMS;          // [16][512][128] new_mom

  unsigned short* ampF  = (unsigned short*)d_ws;                        // 2 MB bf16
  float*          pre   = (float*)((char*)d_ws + (2u << 20));           // 2 MB f32
  unsigned int*   maxbf = (unsigned int*)((char*)d_ws + (4u << 20));    // 64 B

  frames_kernel<<<32, 256, 0, stream>>>(x, Wi, bi, Wm, bm, nm_out, ampF, maxbf);
  chunk_gemm_kernel<<<N_FRAMES, 256, 0, stream>>>(atoms, ampF, pre, maxbf);
  normalize_kernel<<<(WINDOWED_ELEMS + 255) / 256, 256, 0, stream>>>(pre, maxbf, out);
}
